// RecurrenceInjector_22668837388383
// MI455X (gfx1250) — compile-verified, matrix operand order checked by reading, not run on hardware
//
#include <hip/hip_runtime.h>
#include <hip/hip_bf16.h>

// Problem constants (match reference)
#define S_LEN   2048
#define D_DIM   1024
#define H_NUM   16
#define DH_DIM  64
#define B_NUM   4
#define M_ROWS  (B_NUM * S_LEN)   // 8192

#define LOG2E   1.4426950408889634f
#define QSCALE  (0.125f * LOG2E)   // 1/sqrt(DH) * log2(e), folded into Q

typedef __bf16 bf16;
typedef __attribute__((ext_vector_type(16))) __bf16 bf16x16;
typedef __attribute__((ext_vector_type(8)))  float  f32x8;

// ---- CDNA5 async global->LDS copy (ASYNCcnt-tracked), with safe fallback ----
#if __has_builtin(__builtin_amdgcn_global_load_async_to_lds_b128) && \
    __has_builtin(__builtin_amdgcn_s_wait_asynccnt)
#define HAS_ASYNC_LDS 1
#else
#define HAS_ASYNC_LDS 0
#endif

typedef int v4i __attribute__((vector_size(16)));
typedef __attribute__((address_space(1))) v4i gv4i;   // global (AS1)
typedef __attribute__((address_space(3))) v4i lv4i;   // LDS (AS3)

__device__ __forceinline__ void async_copy16(const void* g, void* l) {
#if HAS_ASYNC_LDS
    __builtin_amdgcn_global_load_async_to_lds_b128((gv4i*)g, (lv4i*)l, 0, 0);
#else
    *(uint4*)l = *(const uint4*)g;
#endif
}
__device__ __forceinline__ void async_wait_all() {
#if HAS_ASYNC_LDS
    __builtin_amdgcn_s_wait_asynccnt(0);
#endif
}

// hardware f32 -> bf16 convert (RNE); backend pairs these into packed cvts
__device__ __forceinline__ bf16 f2bf(float f) { return (bf16)f; }

// Load a 16-element bf16 WMMA fragment for one lane.
// p points at (row_base + 8*kHalf); elements [0..7] = p[0..7] (K = 8*kHalf+e),
// [8..15] = p[16..23] (K = 16+8*kHalf+e-8) -- CDNA5 wave32 16-bit operand layout.
__device__ __forceinline__ bf16x16 load_frag_row(const bf16* p) {
    union { bf16x16 v; uint4 q[2]; } u;
    u.q[0] = *(const uint4*)(p);
    u.q[1] = *(const uint4*)(p + 16);
    return u.v;
}

__device__ __forceinline__ f32x8 wmma_bf16(bf16x16 a, bf16x16 b, f32x8 c) {
    return __builtin_amdgcn_wmma_f32_16x16x32_bf16(
        false, a, false, b, (short)0, c, false, false);
}

// ---------------------------------------------------------------------------
// Stage 0a: cast x / previous_state f32 -> bf16
// ---------------------------------------------------------------------------
__global__ __launch_bounds__(256)
void pack_inputs(const float* __restrict__ x, const float* __restrict__ p,
                 bf16* __restrict__ xb, bf16* __restrict__ pb, int n) {
    int i = blockIdx.x * 256 + threadIdx.x;
    if (i < n) {
        xb[i] = f2bf(x[i]);
        pb[i] = f2bf(p[i]);
    }
}

// ---------------------------------------------------------------------------
// Stage 0b: transpose+cast a DxD weight: Wt[n][k] = W[k][n]  (bf16)
// ---------------------------------------------------------------------------
__global__ __launch_bounds__(256)
void transpose_cast(const float* __restrict__ W, bf16* __restrict__ Wt) {
    __shared__ float tile[32][33];
    const int tx = threadIdx.x;          // 0..31
    const int ty = threadIdx.y;          // 0..7
    const int n0 = blockIdx.x * 32;
    const int k0 = blockIdx.y * 32;
    #pragma unroll
    for (int r = 0; r < 32; r += 8)
        tile[ty + r][tx] = W[(size_t)(k0 + ty + r) * D_DIM + n0 + tx];
    __syncthreads();
    #pragma unroll
    for (int r = 0; r < 32; r += 8)
        Wt[(size_t)(n0 + ty + r) * D_DIM + k0 + tx] = f2bf(tile[tx][ty + r]);
}

// ---------------------------------------------------------------------------
// Stage 1/3: bf16 GEMM  C[M,N] = A[M,K] * Wt[N,K]^T + bias, fused epilogues.
// 256 thr = 8 waves; tile 128x128x32; double-buffered LDS filled by async
// global->LDS copies; each wave computes a 64x32 strip = 4x2 WMMA tiles.
// ---------------------------------------------------------------------------
#define BM 128
#define BN 128
#define BK 32
#define LDT 40   // padded LDS row stride (elements); 80B keeps 16B alignment

enum { MODE_Q = 0, MODE_K = 1, MODE_V = 2, MODE_O = 3 };

template <int MODE>
__global__ __launch_bounds__(256, 1)
void gemm_wmma(const bf16* __restrict__ A, const bf16* __restrict__ Bt,
               const float* __restrict__ bias,
               bf16* __restrict__ outh,
               const float* __restrict__ xres,
               const int* __restrict__ mask,
               float* __restrict__ outf) {
    __shared__ bf16 As[2][BM * LDT];
    __shared__ bf16 Bs[2][BN * LDT];

    const int tid   = threadIdx.x;
    const int lane  = tid & 31;
    const int wid   = tid >> 5;        // 0..7
    const int wm    = wid >> 2;        // 0..1 -> 64-row strip
    const int wn    = wid & 3;         // 0..3 -> 32-col strip
    const int klane = lane >> 4;       // K-half select
    const int lr    = lane & 15;       // row/col within 16

    const int n0 = blockIdx.x * BN;
    const int m0 = blockIdx.y * BM;

    // this thread's two 16B staging chunks per tile
    int rowc[2], kcc[2];
    #pragma unroll
    for (int c = 0; c < 2; ++c) {
        const int ch = tid + c * 256;
        rowc[c] = ch >> 2;
        kcc[c]  = (ch & 3) * 8;
    }

    f32x8 acc[4][2];
    #pragma unroll
    for (int mt = 0; mt < 4; ++mt)
        #pragma unroll
        for (int nt = 0; nt < 2; ++nt)
            #pragma unroll
            for (int e = 0; e < 8; ++e) acc[mt][nt][e] = 0.0f;

    // preload K-step 0 into buffer 0 (async fire-and-forget)
    #pragma unroll
    for (int c = 0; c < 2; ++c) {
        async_copy16(&A[(size_t)(m0 + rowc[c]) * D_DIM + kcc[c]],
                     &As[0][rowc[c] * LDT + kcc[c]]);
        async_copy16(&Bt[(size_t)(n0 + rowc[c]) * D_DIM + kcc[c]],
                     &Bs[0][rowc[c] * LDT + kcc[c]]);
    }

    int cur = 0;
    for (int kb = 0; kb < D_DIM; kb += BK) {
        async_wait_all();                      // our fills of buf[cur] landed
        __syncthreads();                       // everyone's fills landed
        const bool nxt = (kb + BK) < D_DIM;
        if (nxt) {                             // fire next tile into other buf
            #pragma unroll
            for (int c = 0; c < 2; ++c) {
                async_copy16(&A[(size_t)(m0 + rowc[c]) * D_DIM + kb + BK + kcc[c]],
                             &As[cur ^ 1][rowc[c] * LDT + kcc[c]]);
                async_copy16(&Bt[(size_t)(n0 + rowc[c]) * D_DIM + kb + BK + kcc[c]],
                             &Bs[cur ^ 1][rowc[c] * LDT + kcc[c]]);
            }
        }

        bf16x16 af[4], bfr[2];
        #pragma unroll
        for (int mt = 0; mt < 4; ++mt)
            af[mt] = load_frag_row(&As[cur][(wm * 64 + mt * 16 + lr) * LDT + 8 * klane]);
        #pragma unroll
        for (int nt = 0; nt < 2; ++nt)
            bfr[nt] = load_frag_row(&Bs[cur][(wn * 32 + nt * 16 + lr) * LDT + 8 * klane]);

        #pragma unroll
        for (int mt = 0; mt < 4; ++mt)
            #pragma unroll
            for (int nt = 0; nt < 2; ++nt)
                acc[mt][nt] = wmma_bf16(af[mt], bfr[nt], acc[mt][nt]);

        if (nxt) cur ^= 1;
    }

    // Epilogue. C/D layout: lane lr -> N, VGPR e -> M = e + 8*klane.
    #pragma unroll
    for (int mt = 0; mt < 4; ++mt) {
        #pragma unroll
        for (int nt = 0; nt < 2; ++nt) {
            #pragma unroll
            for (int e = 0; e < 8; ++e) {
                const int m = m0 + wm * 64 + mt * 16 + klane * 8 + e;
                const int n = n0 + wn * 32 + nt * 16 + lr;
                float v = acc[mt][nt][e] + bias[n];
                if (MODE == MODE_O) {
                    const int b = m >> 11;  // m / S_LEN
                    const float g = (mask[b] != 0) ? 1.0f : 0.0f;
                    outf[(size_t)m * D_DIM + n] =
                        xres[(size_t)m * D_DIM + n] + g * v;
                } else {
                    const int b = m >> 11, s = m & (S_LEN - 1);
                    const int h = n >> 6, d = n & (DH_DIM - 1);
                    if (MODE == MODE_Q) {
                        // fold 1/sqrt(DH) * log2(e): softmax runs in base-2
                        outh[((size_t)(b * H_NUM + h) * S_LEN + s) * DH_DIM + d] =
                            f2bf(v * QSCALE);
                    } else if (MODE == MODE_K) {
                        outh[((size_t)(b * H_NUM + h) * S_LEN + s) * DH_DIM + d] =
                            f2bf(v);
                    } else {                      // MODE_V: transposed [b,h,d,s]
                        outh[((size_t)(b * H_NUM + h) * DH_DIM + d) * S_LEN + s] =
                            f2bf(v);
                    }
                }
            }
        }
    }
}

// ---------------------------------------------------------------------------
// Stage 2: flash attention per (b, h); block = 4 waves x 16 q-rows.
// Double-buffered K/V tiles filled by async global->LDS copies; softmax in
// A-layout and in the log2 domain (exp2 = bare v_exp_f32); O-rescale skipped
// when the running max is stable.
// ---------------------------------------------------------------------------
#define KLD 72   // K-tile LDS row stride (elems): 64 + 8 pad = 144B
#define VLD 40   // V-tile LDS row stride (elems): 32 + 8 pad = 80B
#define SLD 36   // score bounce row stride (floats): 32 + 4 pad = 144B

__global__ __launch_bounds__(128, 1)
void attn_wmma(const bf16* __restrict__ Qh, const bf16* __restrict__ Kh,
               const bf16* __restrict__ Vt, bf16* __restrict__ Oh) {
    __shared__ bf16  Ks[2][32 * KLD];     // 9 KB,  shared by all 4 waves
    __shared__ bf16  Vs[2][64 * VLD];     // 10 KB, shared by all 4 waves
    __shared__ float Sf[4][16 * SLD];     // 9 KB,  per-wave score bounce
    __shared__ float Cf[4][16];           // per-wave row-broadcast line

    const int tid   = threadIdx.x;
    const int lane  = tid & 31;
    const int wid   = tid >> 5;        // 0..3
    const int klane = lane >> 4;
    const int lr    = lane & 15;

    const int b  = blockIdx.z;
    const int h  = blockIdx.y;
    const int q0 = blockIdx.x * 64 + wid * 16;

    const size_t bh = (size_t)(b * H_NUM + h);
    const bf16* Qbase = Qh + (bh * S_LEN + q0) * DH_DIM;
    const bf16* Kbase = Kh + bh * S_LEN * DH_DIM;
    const bf16* Vbase = Vt + bh * DH_DIM * S_LEN;

    // per-thread staging chunks (256 x 16B per tile, 2 per thread)
    int krow[2], kdc[2], vrow[2], vkc[2];
    #pragma unroll
    for (int c = 0; c < 2; ++c) {
        const int ch = tid + c * 128;
        krow[c] = ch >> 3;  kdc[c] = (ch & 7) * 8;
        vrow[c] = ch >> 2;  vkc[c] = (ch & 3) * 8;
    }

    // preload kv-tile 0 into buffer 0
    #pragma unroll
    for (int c = 0; c < 2; ++c) {
        async_copy16(&Kbase[(size_t)krow[c] * DH_DIM + kdc[c]],
                     &Ks[0][krow[c] * KLD + kdc[c]]);
        async_copy16(&Vbase[(size_t)vrow[c] * S_LEN + vkc[c]],
                     &Vs[0][vrow[c] * VLD + vkc[c]]);
    }

    // Q A-fragments (row = lr), kept in VGPRs for the whole kv sweep
    bf16x16 aq0 = load_frag_row(Qbase + (size_t)lr * DH_DIM + 0  + 8 * klane);
    bf16x16 aq1 = load_frag_row(Qbase + (size_t)lr * DH_DIM + 32 + 8 * klane);

    f32x8 o[4];
    #pragma unroll
    for (int t = 0; t < 4; ++t)
        #pragma unroll
        for (int e = 0; e < 8; ++e) o[t][e] = 0.0f;
    float mrow = -3e38f;   // stats for row (lane&15), replicated in both halves
    float lrow = 0.0f;

    // alibi slope in the log2 domain: 2^(-8*(h+1)/H) * log2(e)
    const float slope = exp2f(-0.5f * (float)(h + 1)) * LOG2E;
    const float fi0 = (float)(q0 + klane * 8);           // row base (float)
    const float fj0 = (float)(lr);                       // col bases (float)
    const float fj1 = (float)(16 + lr);
    float kvf = 0.0f;

    int cur = 0;
    for (int kv0 = 0; kv0 < S_LEN; kv0 += 32) {
        async_wait_all();                 // fills of buf[cur] by this wave done
        __syncthreads();                  // fills by all waves done
        const bool nxt = (kv0 + 32) < S_LEN;
        if (nxt) {                        // fire next kv tile into other buffer
            #pragma unroll
            for (int c = 0; c < 2; ++c) {
                async_copy16(&Kbase[(size_t)(kv0 + 32 + krow[c]) * DH_DIM + kdc[c]],
                             &Ks[cur ^ 1][krow[c] * KLD + kdc[c]]);
                async_copy16(&Vbase[(size_t)vrow[c] * S_LEN + kv0 + 32 + vkc[c]],
                             &Vs[cur ^ 1][vrow[c] * VLD + vkc[c]]);
            }
        }

        // ---- scores: two 16x16 C tiles, contraction over DH=64
        f32x8 c[2];
        #pragma unroll
        for (int nt = 0; nt < 2; ++nt) {
            bf16x16 k0 = load_frag_row(&Ks[cur][(nt * 16 + lr) * KLD + 0  + 8 * klane]);
            bf16x16 k1 = load_frag_row(&Ks[cur][(nt * 16 + lr) * KLD + 32 + 8 * klane]);
            f32x8 cc;
            #pragma unroll
            for (int e = 0; e < 8; ++e) cc[e] = 0.0f;
            cc = wmma_bf16(aq0, k0, cc);
            cc = wmma_bf16(aq1, k1, cc);
            c[nt] = cc;
        }
        // ---- alibi bias, all-float (no int->float converts in the loop)
        {
            const float j0 = fj0 + kvf, j1 = fj1 + kvf;
            #pragma unroll
            for (int e = 0; e < 8; ++e) {
                const float fi = fi0 + (float)e;
                c[0][e] -= slope * fabsf(fi - j0);
                c[1][e] -= slope * fabsf(fi - j1);
            }
            kvf += 32.0f;
        }

        // ---- bounce raw scores to LDS (C-layout positions -> row-major)
        float* sf = &Sf[wid][0];
        #pragma unroll
        for (int nt = 0; nt < 2; ++nt)
            #pragma unroll
            for (int e = 0; e < 8; ++e)
                sf[(klane * 8 + e) * SLD + nt * 16 + lr] = c[nt][e];
        asm volatile("" ::: "memory");

        // ---- read back in A-layout: this lane holds 16 values of row lr
        const float* sp = &Sf[wid][lr * SLD + 8 * klane];
        float s[16];
        { float4 a0 = *(const float4*)(sp);
          float4 a1 = *(const float4*)(sp + 4);
          float4 a2 = *(const float4*)(sp + 16);
          float4 a3 = *(const float4*)(sp + 20);
          s[0]=a0.x; s[1]=a0.y; s[2]=a0.z; s[3]=a0.w;
          s[4]=a1.x; s[5]=a1.y; s[6]=a1.z; s[7]=a1.w;
          s[8]=a2.x; s[9]=a2.y; s[10]=a2.z; s[11]=a2.w;
          s[12]=a3.x; s[13]=a3.y; s[14]=a3.z; s[15]=a3.w; }

        // ---- online softmax, A-layout, base-2
        float tmax = s[0];
        #pragma unroll
        for (int i = 1; i < 16; ++i) tmax = fmaxf(tmax, s[i]);
        tmax = fmaxf(tmax, __shfl_xor(tmax, 16, 32));

        const bool upd = __any(tmax > mrow);   // wave-uniform: EXEC stays full
        float corr = 1.0f;
        if (upd) {
            const float mnew = fmaxf(mrow, tmax);
            corr = exp2f(mrow - mnew);
            mrow = mnew;
        }
        float p[16], rs = 0.0f;
        #pragma unroll
        for (int i = 0; i < 16; ++i) { p[i] = exp2f(s[i] - mrow); rs += p[i]; }
        rs += __shfl_xor(rs, 16, 32);

        bf16x16 pa;
        #pragma unroll
        for (int i = 0; i < 16; ++i) pa[i] = f2bf(p[i]);

        if (upd) {
            lrow = lrow * corr + rs;
            // broadcast corr into C-layout (row e+8*klane per lane)
            if (klane == 0) Cf[wid][lr] = corr;
            asm volatile("" ::: "memory");
            const float* cp = &Cf[wid][klane * 8];
            float4 ca = *(const float4*)(cp);
            float4 cb = *(const float4*)(cp + 4);
            const float ce[8] = {ca.x, ca.y, ca.z, ca.w, cb.x, cb.y, cb.z, cb.w};
            #pragma unroll
            for (int t4 = 0; t4 < 4; ++t4)
                #pragma unroll
                for (int e = 0; e < 8; ++e) o[t4][e] *= ce[e];
        } else {
            lrow += rs;
        }

        // ---- PV: 4 output d-tiles, contraction over this kv block (K=32)
        #pragma unroll
        for (int t4 = 0; t4 < 4; ++t4) {
            bf16x16 vf = load_frag_row(&Vs[cur][(t4 * 16 + lr) * VLD + 8 * klane]);
            o[t4] = wmma_bf16(pa, vf, o[t4]);
        }

        if (nxt) cur ^= 1;
    }

    // ---- epilogue: O /= l (broadcast 1/l into C-layout), write [B,S,D]
    if (klane == 0) Cf[wid][lr] = 1.0f / lrow;
    asm volatile("" ::: "memory");
    const float* ip = &Cf[wid][klane * 8];
    float4 ia = *(const float4*)(ip);
    float4 ib = *(const float4*)(ip + 4);
    const float inv[8] = {ia.x, ia.y, ia.z, ia.w, ib.x, ib.y, ib.z, ib.w};
    #pragma unroll
    for (int e = 0; e < 8; ++e) {
        const int srow = q0 + klane * 8 + e;
        const size_t base = ((size_t)b * S_LEN + srow) * D_DIM + h * DH_DIM;
        #pragma unroll
        for (int t4 = 0; t4 < 4; ++t4)
            Oh[base + t4 * 16 + lr] = f2bf(o[t4][e] * inv[e]);
    }
}

// ---------------------------------------------------------------------------
extern "C" void kernel_launch(void* const* d_in, const int* in_sizes, int n_in,
                              void* d_out, int out_size, void* d_ws, size_t ws_size,
                              hipStream_t stream) {
    const float* x    = (const float*)d_in[0];
    const float* prev = (const float*)d_in[1];
    const float* Wq   = (const float*)d_in[2];
    const float* Wk   = (const float*)d_in[3];
    const float* Wv   = (const float*)d_in[4];
    const float* Wo   = (const float*)d_in[5];
    const float* bq   = (const float*)d_in[6];
    const float* bk   = (const float*)d_in[7];
    const float* bv   = (const float*)d_in[8];
    const float* bo   = (const float*)d_in[9];
    const int*   mask = (const int*)d_in[10];
    float* out = (float*)d_out;

    char* ws = (char*)d_ws;
    const size_t MB = 1024ull * 1024ull;
    bf16* xb  = (bf16*)(ws);              // 16 MB, reused as attn-out later
    bf16* pb  = (bf16*)(ws + 16 * MB);    // 16 MB
    bf16* WqT = (bf16*)(ws + 32 * MB);    // 2 MB each
    bf16* WkT = (bf16*)(ws + 34 * MB);
    bf16* WvT = (bf16*)(ws + 36 * MB);
    bf16* WoT = (bf16*)(ws + 38 * MB);
    bf16* Qh  = (bf16*)(ws + 40 * MB);    // 16 MB [B,H,S,DH]
    bf16* Kh  = (bf16*)(ws + 56 * MB);    // 16 MB [B,H,S,DH]
    bf16* Vt  = (bf16*)(ws + 72 * MB);    // 16 MB [B,H,DH,S]
    bf16* Ah  = xb;                       // attention output [B,S,D], reuses xb

    const int ntok = M_ROWS * D_DIM;      // 8,388,608
    pack_inputs<<<dim3((ntok + 255) / 256), dim3(256), 0, stream>>>(x, prev, xb, pb, ntok);

    dim3 tb(32, 8), tg(D_DIM / 32, D_DIM / 32);
    transpose_cast<<<tg, tb, 0, stream>>>(Wq, WqT);
    transpose_cast<<<tg, tb, 0, stream>>>(Wk, WkT);
    transpose_cast<<<tg, tb, 0, stream>>>(Wv, WvT);
    transpose_cast<<<tg, tb, 0, stream>>>(Wo, WoT);

    dim3 gg(D_DIM / BN, M_ROWS / BM);     // (8, 64)
    gemm_wmma<MODE_Q><<<gg, dim3(256), 0, stream>>>(xb, WqT, bq, Qh, nullptr, nullptr, nullptr);
    gemm_wmma<MODE_K><<<gg, dim3(256), 0, stream>>>(pb, WkT, bk, Kh, nullptr, nullptr, nullptr);
    gemm_wmma<MODE_V><<<gg, dim3(256), 0, stream>>>(pb, WvT, bv, Vt, nullptr, nullptr, nullptr);

    attn_wmma<<<dim3(S_LEN / 64, H_NUM, B_NUM), dim3(128), 0, stream>>>(Qh, Kh, Vt, Ah);

    gemm_wmma<MODE_O><<<gg, dim3(256), 0, stream>>>(Ah, WoT, bo, nullptr, x, mask, out);
}
